// HybridFNN_16097537425654
// MI455X (gfx1250) — compile-verified, matrix operand order checked by reading, not run on hardware
//
#include <hip/hip_runtime.h>
#include <stdint.h>

#define N_CHN 95
#define GDIM  4
#define XSTR  (GDIM + 2*N_CHN)   // 194 floats per input row
#define GH    128
#define CHH   32
#define TILES_A 2                // row-tiles per wave, g-MLP kernel
#define TILES_C 4                // row-tiles per wave, output kernel

typedef __attribute__((ext_vector_type(16))) __bf16 v16bf;
typedef __attribute__((ext_vector_type(2)))  __bf16 v2bf;
typedef __attribute__((ext_vector_type(8)))  float  v8f;

union Frag {
  v16bf v;
  uint4 q[2];
  unsigned u[8];
};

// f32 -> bf16 via hardware converts (RNE); pair packed into one dword (K-pair)
__device__ __forceinline__ unsigned short bf16one(float a) {
  __bf16 h = (__bf16)a;
  return __builtin_bit_cast(unsigned short, h);
}
__device__ __forceinline__ unsigned bf16pair(float a, float b) {
  v2bf p = { (__bf16)a, (__bf16)b };
  return __builtin_bit_cast(unsigned, p);
}
__device__ __forceinline__ v8f wmma_bf16(const Frag& a, const Frag& b, v8f c) {
  return __builtin_amdgcn_wmma_f32_16x16x32_bf16(false, a.v, false, b.v,
                                                 (short)0, c, false, false);
}

// ---------------------------------------------------------------------------
// Kernel A: global MLP 4->128->128->128, output g_emb as bf16 (B x 128)
// 256 threads = 8 waves; each wave sweeps TILES_A 16-row batch tiles.
// ---------------------------------------------------------------------------
__global__ __launch_bounds__(256) void g_mlp_kernel(
    const float* __restrict__ x,
    const float* __restrict__ gW1, const float* __restrict__ gb1,
    const float* __restrict__ gW2, const float* __restrict__ gb2,
    const float* __restrict__ gW3, const float* __restrict__ gb3,
    unsigned short* __restrict__ gemb, int Btot)
{
  __shared__ __align__(16) float sW1[4*GH];
  __shared__ float sB1[GH], sB2[GH], sB3[GH];
  __shared__ __align__(16) unsigned sW2f[32*256];   // 32 B-tiles (32x16) frag dwords
  __shared__ __align__(16) unsigned sW3f[32*256];
  __shared__ __align__(16) unsigned short sStage[8][16*GH]; // per-wave 16x128 bf16

  const int tid = threadIdx.x;
  for (int i = tid; i < 4*GH; i += 256) sW1[i] = gW1[i];
  for (int i = tid; i < GH; i += 256) { sB1[i]=gb1[i]; sB2[i]=gb2[i]; sB3[i]=gb3[i]; }
  // build B fragments: tile t = kk*8+nt; lane ln; dword j packs K-pair
  for (int idx = tid; idx < 32*256; idx += 256) {
    int t  = idx >> 8;
    int r  = idx & 255;
    int ln = r >> 3, j = r & 7;
    int half = ln >> 4;
    int n  = (t & 7)*16 + (ln & 15);
    int k0 = (t >> 3)*32 + ((j >= 4) ? 16 : 0) + half*8 + (j & 3)*2;
    sW2f[idx] = bf16pair(gW2[k0*GH + n], gW2[(k0+1)*GH + n]);
    sW3f[idx] = bf16pair(gW3[k0*GH + n], gW3[(k0+1)*GH + n]);
  }
  __syncthreads();

  const int lane = tid & 31, w = tid >> 5;
  const int l = lane & 15, half = lane >> 4;

  #pragma unroll 1
  for (int t = 0; t < TILES_A; ++t) {
    const int m0 = blockIdx.x*(128*TILES_A) + t*128 + w*16;
    int b = m0 + l; if (b >= Btot) b = Btot - 1;

    float2 x01 = *(const float2*)(x + (size_t)b*XSTR);
    float2 x23 = *(const float2*)(x + (size_t)b*XSTR + 2);

    // layer 1 (K=4) computed straight into A-fragment layout
    Frag a[4];
    #pragma unroll
    for (int kk = 0; kk < 4; ++kk) {
      #pragma unroll
      for (int j = 0; j < 8; ++j) {
        int h0 = kk*32 + ((j>=4)?16:0) + half*8 + (j&3)*2;
        int h1 = h0 + 1;
        float v0 = x01.x*sW1[h0] + x01.y*sW1[GH+h0] + x23.x*sW1[2*GH+h0] + x23.y*sW1[3*GH+h0] + sB1[h0];
        float v1 = x01.x*sW1[h1] + x01.y*sW1[GH+h1] + x23.x*sW1[2*GH+h1] + x23.y*sW1[3*GH+h1] + sB1[h1];
        a[kk].u[j] = bf16pair(fmaxf(v0, 0.f), fmaxf(v1, 0.f));
      }
    }

    // layer 2: 8 N-tiles x 4 K-steps of WMMA, relu+bias, restage as bf16 rows
    #pragma unroll
    for (int nt = 0; nt < 8; ++nt) {
      v8f acc = {};
      #pragma unroll
      for (int kk = 0; kk < 4; ++kk) {
        Frag bb;
        const unsigned* src = &sW2f[(kk*8 + nt)*256 + lane*8];
        bb.q[0] = *(const uint4*)src;
        bb.q[1] = *(const uint4*)(src + 4);
        acc = wmma_bf16(a[kk], bb, acc);
      }
      int col = nt*16 + l;
      #pragma unroll
      for (int v = 0; v < 8; ++v) {
        float val = fmaxf(acc[v] + sB2[col], 0.f);
        sStage[w][(v + 8*half)*GH + col] = bf16one(val);
      }
    }
    asm volatile("s_wait_dscnt 0" ::: "memory");   // intra-wave LDS RAW
    #pragma unroll
    for (int kk = 0; kk < 4; ++kk) {
      const unsigned short* src = &sStage[w][l*GH + kk*32 + half*8];
      a[kk].q[0] = *(const uint4*)src;
      a[kk].q[1] = *(const uint4*)(src + 16);
    }

    // layer 3: no relu, store g_emb bf16 row-major (contiguous 32B per lane-half)
    #pragma unroll
    for (int nt = 0; nt < 8; ++nt) {
      v8f acc = {};
      #pragma unroll
      for (int kk = 0; kk < 4; ++kk) {
        Frag bb;
        const unsigned* src = &sW3f[(kk*8 + nt)*256 + lane*8];
        bb.q[0] = *(const uint4*)src;
        bb.q[1] = *(const uint4*)(src + 4);
        acc = wmma_bf16(a[kk], bb, acc);
      }
      int col = nt*16 + l;
      #pragma unroll
      for (int v = 0; v < 8; ++v) {
        int row = m0 + v + 8*half;
        if (row < Btot) gemb[(size_t)row*GH + col] = bf16one(acc[v] + sB3[col]);
      }
    }
  }
}

// ---------------------------------------------------------------------------
// Kernel C: per (channel, 512-row slab) block; each wave sweeps TILES_C tiles.
// c-MLP layer1 on VALU -> A-frag; c_emb via WMMA (K=32);
// main h = relu(g_emb@oW1_g + c_emb@oW1_c + ob1) via 2 N-tiles x 5 WMMA;
// fused 32->1 dot via xor-shuffle reduction.
// ---------------------------------------------------------------------------
__global__ __launch_bounds__(256) void out_kernel(
    const float* __restrict__ x,
    const float* __restrict__ cW1, const float* __restrict__ cb1,
    const float* __restrict__ cW2, const float* __restrict__ cb2,
    const float* __restrict__ oW1, const float* __restrict__ ob1,
    const float* __restrict__ oW2, const float* __restrict__ ob2,
    const unsigned short* __restrict__ gemb,
    float* __restrict__ out, int Btot)
{
  __shared__ __align__(16) unsigned sBg[8*256];   // oW1_g: 4 K-steps x 2 N-tiles
  __shared__ __align__(16) unsigned sBc[2*256];   // oW1_c: 2 N-tiles
  __shared__ __align__(16) unsigned sBw2[2*256];  // cW2:   2 N-tiles
  __shared__ float sCW1[2*CHH], sCb1[CHH], sCb2[CHH], sOb1[CHH], sOw2[CHH];
  __shared__ float sOb2;
  __shared__ __align__(16) unsigned short sCe[8][16*CHH]; // per-wave c_emb tile

  const int ch  = blockIdx.x;
  const int tid = threadIdx.x;
  const float* oW1ch = oW1 + (size_t)ch*(GH+CHH)*CHH;  // 160x32 f32

  // pull this channel's weight tile toward the caches (global_prefetch_b8)
  __builtin_prefetch(oW1ch + tid*20, 0, 0);

  for (int idx = tid; idx < 8*256; idx += 256) {
    int t = idx >> 8;                       // kk*2 + nt
    int r = idx & 255, ln = r >> 3, j = r & 7;
    int half = ln >> 4;
    int n  = (t & 1)*16 + (ln & 15);
    int k0 = (t >> 1)*32 + ((j>=4)?16:0) + half*8 + (j&3)*2;
    sBg[idx] = bf16pair(oW1ch[k0*CHH + n], oW1ch[(k0+1)*CHH + n]);
  }
  for (int idx = tid; idx < 2*256; idx += 256) {
    int t = idx >> 8;                       // nt
    int r = idx & 255, ln = r >> 3, j = r & 7;
    int half = ln >> 4;
    int n  = t*16 + (ln & 15);
    int k0 = ((j>=4)?16:0) + half*8 + (j&3)*2;
    sBc[idx]  = bf16pair(oW1ch[(GH + k0)*CHH + n], oW1ch[(GH + k0 + 1)*CHH + n]);
    sBw2[idx] = bf16pair(cW2[k0*CHH + n], cW2[(k0+1)*CHH + n]);
  }
  if (tid < CHH) {
    sCb1[tid] = cb1[tid]; sCb2[tid] = cb2[tid];
    sOb1[tid] = ob1[ch*CHH + tid]; sOw2[tid] = oW2[ch*CHH + tid];
  }
  if (tid < 2*CHH) sCW1[tid] = cW1[tid];
  if (tid == 0)    sOb2 = ob2[ch];
  __syncthreads();

  const int lane = tid & 31, w = tid >> 5;
  const int l = lane & 15, half = lane >> 4;

  #pragma unroll 1
  for (int t = 0; t < TILES_C; ++t) {
    const int m0 = blockIdx.y*(128*TILES_C) + t*128 + w*16;
    int b = m0 + l; if (b >= Btot) b = Btot - 1;

    // A fragments of g_emb (16x128 -> 4 tiles), straight from global bf16 rows
    Frag ag[4];
    const unsigned short* grow = gemb + (size_t)b*GH;
    #pragma unroll
    for (int kk = 0; kk < 4; ++kk) {
      ag[kk].q[0] = *(const uint4*)(grow + kk*32 + half*8);
      ag[kk].q[1] = *(const uint4*)(grow + kk*32 + 16 + half*8);
    }

    // channel layer-1 (K=2) on VALU, written directly in A-fragment layout
    float2 sw = *(const float2*)(x + (size_t)b*XSTR + GDIM + 2*ch);
    float sp = sw.x * 0.01f, wsv = sw.y;
    Frag ac;
    #pragma unroll
    for (int j = 0; j < 8; ++j) {
      int h0 = ((j>=4)?16:0) + half*8 + (j&3)*2, h1 = h0 + 1;
      float v0 = fmaxf(sp*sCW1[h0] + wsv*sCW1[CHH+h0] + sCb1[h0], 0.f);
      float v1 = fmaxf(sp*sCW1[h1] + wsv*sCW1[CHH+h1] + sCb1[h1], 0.f);
      ac.u[j] = bf16pair(v0, v1);
    }

    // c_emb = ac @ cW2 + cb2 -> restage to LDS (row-major bf16 16x32)
    #pragma unroll
    for (int nt = 0; nt < 2; ++nt) {
      v8f acc = {};
      Frag bb;
      const unsigned* src = &sBw2[nt*256 + lane*8];
      bb.q[0] = *(const uint4*)src; bb.q[1] = *(const uint4*)(src + 4);
      acc = wmma_bf16(ac, bb, acc);
      int col = nt*16 + l;
      float cb = sCb2[col];
      #pragma unroll
      for (int v = 0; v < 8; ++v)
        sCe[w][(v + 8*half)*CHH + col] = bf16one(acc[v] + cb);
    }
    asm volatile("s_wait_dscnt 0" ::: "memory");
    Frag ace;
    {
      const unsigned short* src = &sCe[w][l*CHH + half*8];
      ace.q[0] = *(const uint4*)src;
      ace.q[1] = *(const uint4*)(src + 16);
    }

    // main GEMM (K=160) + fused relu/bias + 32->1 dot
    float partial[8];
    #pragma unroll
    for (int v = 0; v < 8; ++v) partial[v] = 0.f;
    #pragma unroll
    for (int nt = 0; nt < 2; ++nt) {
      v8f acc = {};
      #pragma unroll
      for (int kk = 0; kk < 4; ++kk) {
        Frag bb;
        const unsigned* src = &sBg[(kk*2 + nt)*256 + lane*8];
        bb.q[0] = *(const uint4*)src; bb.q[1] = *(const uint4*)(src + 4);
        acc = wmma_bf16(ag[kk], bb, acc);
      }
      {
        Frag bb;
        const unsigned* src = &sBc[nt*256 + lane*8];
        bb.q[0] = *(const uint4*)src; bb.q[1] = *(const uint4*)(src + 4);
        acc = wmma_bf16(ace, bb, acc);
      }
      int col = nt*16 + l;
      float bias = sOb1[col], w2 = sOw2[col];
      #pragma unroll
      for (int v = 0; v < 8; ++v)
        partial[v] += fmaxf(acc[v] + bias, 0.f) * w2;
    }
    // reduce over the 16 N-lanes of each half
    #pragma unroll
    for (int v = 0; v < 8; ++v) {
      partial[v] += __shfl_xor(partial[v], 1, 32);
      partial[v] += __shfl_xor(partial[v], 2, 32);
      partial[v] += __shfl_xor(partial[v], 4, 32);
      partial[v] += __shfl_xor(partial[v], 8, 32);
    }
    if (l == 0) {
      #pragma unroll
      for (int v = 0; v < 8; ++v) {
        int row = m0 + v + 8*half;
        if (row < Btot) out[(size_t)row*N_CHN + ch] = partial[v] + sOb2;
      }
    }
  }
}

// ---------------------------------------------------------------------------
extern "C" void kernel_launch(void* const* d_in, const int* in_sizes, int n_in,
                              void* d_out, int out_size, void* d_ws, size_t ws_size,
                              hipStream_t stream) {
  const float* x   = (const float*)d_in[0];
  const float* gW1 = (const float*)d_in[1];
  const float* gb1 = (const float*)d_in[2];
  const float* gW2 = (const float*)d_in[3];
  const float* gb2 = (const float*)d_in[4];
  const float* gW3 = (const float*)d_in[5];
  const float* gb3 = (const float*)d_in[6];
  const float* cW1 = (const float*)d_in[7];
  const float* cb1 = (const float*)d_in[8];
  const float* cW2 = (const float*)d_in[9];
  const float* cb2 = (const float*)d_in[10];
  const float* oW1 = (const float*)d_in[11];
  const float* ob1 = (const float*)d_in[12];
  const float* oW2 = (const float*)d_in[13];
  const float* ob2 = (const float*)d_in[14];
  float* out = (float*)d_out;

  int Btot = in_sizes[0] / XSTR;
  unsigned short* gemb = (unsigned short*)d_ws;   // B x 128 bf16 (4 MB)

  int nbA = (Btot + 128*TILES_A - 1) / (128*TILES_A);
  g_mlp_kernel<<<nbA, 256, 0, stream>>>(x, gW1, gb1, gW2, gb2, gW3, gb3, gemb, Btot);

  int nbC = (Btot + 128*TILES_C - 1) / (128*TILES_C);
  dim3 grid(N_CHN, nbC);
  out_kernel<<<grid, 256, 0, stream>>>(x, cW1, cb1, cW2, cb2, oW1, ob1, oW2, ob2,
                                       gemb, out, Btot);
}
